// LinearAttention_30562987278514
// MI455X (gfx1250) — matrix operand order repaired and checked by reading, not compiled, not run on hardware
//
#include <hip/hip_runtime.h>

// Problem constants (reference: B=8, C=64, H=W=256)
#define NPIX 65536
#define BATCH 8

typedef __bf16 bf16;
typedef __attribute__((ext_vector_type(16))) __bf16 v16bf;
typedef __attribute__((ext_vector_type(8)))  __bf16 v8bf;
typedef __attribute__((ext_vector_type(8)))  float  v8f;

static __device__ inline v8f wmma_bf16(v16bf a, v16bf b, v8f c) {
  // D(16x16,f32) = A(16x32,bf16) * B(32x16,bf16) + C
  return __builtin_amdgcn_wmma_f32_16x16x32_bf16(
      false, a, false, b, (short)0, c, false, false);
}

// A-fragment loader (16-bit A 16x32), contiguous k order matching B fragments:
// lane half=0 holds K in {kb+0..15}; half=1 holds {kb+16..31}
static __device__ inline v16bf a_frag(const bf16* row, int kbase, int half) {
  v8bf lo = *(const v8bf*)(row + kbase + half * 16);
  v8bf hi = *(const v8bf*)(row + kbase + half * 16 + 8);
  v16bf a;
#pragma unroll
  for (int i = 0; i < 8; ++i) { a[i] = lo[i]; a[i + 8] = hi[i]; }
  return a;
}

static __device__ inline v16bf zero_frag() {
  v16bf z;
#pragma unroll
  for (int i = 0; i < 16; ++i) z[i] = (bf16)0.0f;
  return z;
}

// ---------------------------------------------------------------------------
// Kernel P: transform conv weights (O,I,3,3)f32 -> (dydx,O,I)bf16; zero scratch
// ---------------------------------------------------------------------------
__global__ __launch_bounds__(256) void prep_kernel(
    const float* __restrict__ W1, const float* __restrict__ W2,
    bf16* __restrict__ W1T, bf16* __restrict__ W2T,
    float* __restrict__ zero_base /* MT|Ksum|Vsum: 33792 floats */) {
  int i = blockIdx.x * 256 + threadIdx.x;
  if (i < 36864) {
    int dydx = i >> 12, o = (i >> 6) & 63, c = i & 63;
    W1T[i] = (bf16)W1[(o * 64 + c) * 9 + dydx];
    W2T[i] = (bf16)W2[(o * 64 + c) * 9 + dydx];
  } else {
    int j = i - 36864;
    if (j < 33792) zero_base[j] = 0.0f;
  }
}

// ---------------------------------------------------------------------------
// Kernel 1: QKV projections (1x1 conv = 64x64 GEMM), L2-normalize Q,K.
//   Qn -> (B,N,64) bf16 ;  Kn,V -> (B,64,N) bf16 (via LDS transpose, b128 out)
// ---------------------------------------------------------------------------
__global__ __launch_bounds__(256) void qkv_kernel(
    const float* __restrict__ x,
    const float* __restrict__ Wq, const float* __restrict__ bq,
    const float* __restrict__ Wk, const float* __restrict__ bk,
    const float* __restrict__ Wv, const float* __restrict__ bv,
    bf16* __restrict__ Qn, bf16* __restrict__ Kn, bf16* __restrict__ Vv) {
  __shared__ bf16 Wl[3][64][64];
  __shared__ float bl[3][64];
  __shared__ bf16 kvt[2][64][128];  // [K|V][channel][pixel-in-block]
  int tid = threadIdx.x;
  for (int i = tid; i < 3 * 4096; i += 256) {
    int m = i >> 12, j = i & 4095;
    const float* src = (m == 0) ? Wq : ((m == 1) ? Wk : Wv);
    Wl[m][j >> 6][j & 63] = (bf16)src[j];
  }
  for (int i = tid; i < 3 * 64; i += 256) {
    int m = i >> 6;
    const float* src = (m == 0) ? bq : ((m == 1) ? bk : bv);
    bl[m][i & 63] = src[i & 63];
  }
  __syncthreads();

  int wave = tid >> 5, lane = tid & 31, half = lane >> 4, col = lane & 15;
  long g0 = (long)blockIdx.x * 128 + wave * 16;
  int b = (int)(g0 >> 16);
  int n0 = (int)(g0 & (NPIX - 1));
  int p = n0 + col;

  // B-fragments of x: element e -> channel c0+e, pixel p.
  // Across lanes, each load touches 16 contiguous f32 (coalesced 64B runs).
  v16bf xb[2];
#pragma unroll
  for (int ks = 0; ks < 2; ++ks) {
    int c0 = ks * 32 + half * 16;
    const float* xp = x + ((size_t)b * 64 + c0) * NPIX + p;
#pragma unroll
    for (int e = 0; e < 16; ++e) xb[ks][e] = (bf16)xp[(size_t)e * NPIX];
  }

  v8f acc[3][4];
#pragma unroll
  for (int m = 0; m < 3; ++m)
#pragma unroll
    for (int t = 0; t < 4; ++t)
#pragma unroll
      for (int r = 0; r < 8; ++r) acc[m][t][r] = bl[m][t * 16 + half * 8 + r];

#pragma unroll
  for (int m = 0; m < 3; ++m)
#pragma unroll
    for (int t = 0; t < 4; ++t) {
      int o = t * 16 + col;
#pragma unroll
      for (int ks = 0; ks < 2; ++ks)
        acc[m][t] = wmma_bf16(a_frag(&Wl[m][o][0], ks * 32, half), xb[ks], acc[m][t]);
    }

  // L2 normalize Q (m=0) and K (m=1) along channels (split across lane halves)
#pragma unroll
  for (int m = 0; m < 2; ++m) {
    float s = 0.f;
#pragma unroll
    for (int t = 0; t < 4; ++t)
#pragma unroll
      for (int r = 0; r < 8; ++r) s += acc[m][t][r] * acc[m][t][r];
    s += __shfl_xor(s, 16, 32);
    float inv = rsqrtf(fmaxf(s, 1e-24f));
#pragma unroll
    for (int t = 0; t < 4; ++t)
#pragma unroll
      for (int r = 0; r < 8; ++r) acc[m][t][r] *= inv;
  }

  // Qn: (B,N,64) packed 16B stores (channels t*16+half*8 .. +7 contiguous)
  size_t qbase = ((size_t)b * NPIX + p) * 64;
#pragma unroll
  for (int t = 0; t < 4; ++t) {
    v8bf pk;
#pragma unroll
    for (int r = 0; r < 8; ++r) pk[r] = (bf16)acc[0][t][r];
    *(v8bf*)(Qn + qbase + t * 16 + half * 8) = pk;
  }
  // Kn, V: transpose through LDS, then coalesced b128 stores
  int pl = wave * 16 + col;  // pixel within block [0,128)
#pragma unroll
  for (int t = 0; t < 4; ++t)
#pragma unroll
    for (int r = 0; r < 8; ++r) {
      int c = t * 16 + half * 8 + r;
      kvt[0][c][pl] = (bf16)acc[1][t][r];
      kvt[1][c][pl] = (bf16)acc[2][t][r];
    }
  __syncthreads();
  int nblk = (int)((long)blockIdx.x * 128 & (NPIX - 1));
#pragma unroll
  for (int m = 0; m < 2; ++m) {
    bf16* dst = (m == 0) ? Kn : Vv;
    for (int i = tid; i < 1024; i += 256) {  // 1024 chunks of 8 bf16
      int c = i >> 4, off = (i & 15) * 8;
      v8bf chunk = *(const v8bf*)(&kvt[m][c][off]);
      *(v8bf*)(dst + ((size_t)b * 64 + c) * NPIX + nblk + off) = chunk;
    }
  }
}

// ---------------------------------------------------------------------------
// Kernel 2: MT[vc][kc] = sum_n V[vc,n]*Kn[kc,n]  (per batch), plus
//           Ksum[kc], Vsum[vc]. WMMA with K-dim = pixels; block-level LDS
//           reduction before global f32 atomics.
// ---------------------------------------------------------------------------
__global__ __launch_bounds__(256) void kv_reduce_kernel(
    const bf16* __restrict__ Kn, const bf16* __restrict__ Vv,
    float* __restrict__ MT, float* __restrict__ Ksum, float* __restrict__ Vsum) {
  __shared__ float mred[64][64];
  __shared__ float kred[64], vred[64];
  int tid = threadIdx.x, wave = tid >> 5, lane = tid & 31;
  int half = lane >> 4, col = lane & 15;
  for (int i = tid; i < 4096; i += 256) mred[i >> 6][i & 63] = 0.f;
  if (tid < 64) { kred[tid] = 0.f; vred[tid] = 0.f; }
  __syncthreads();

  long g0 = (long)blockIdx.x * 2048 + wave * 256;
  int b = (int)(g0 >> 16);
  int n0 = (int)(g0 & (NPIX - 1));

  v8f acc[4][4];
#pragma unroll
  for (int vt = 0; vt < 4; ++vt)
#pragma unroll
    for (int kt = 0; kt < 4; ++kt)
#pragma unroll
      for (int r = 0; r < 8; ++r) acc[vt][kt][r] = 0.f;
  float vs[4] = {0.f, 0.f, 0.f, 0.f}, ksm[4] = {0.f, 0.f, 0.f, 0.f};

  for (int it = 0; it < 8; ++it) {
    int pb = n0 + it * 32;
    if (it < 7) {  // prefetch next chunk (global_prefetch)
      __builtin_prefetch(Kn + ((size_t)b * 64 + col) * NPIX + pb + 32, 0, 1);
      __builtin_prefetch(Vv + ((size_t)b * 64 + col) * NPIX + pb + 32, 0, 1);
    }
    v16bf afr[4], bfr[4];
#pragma unroll
    for (int t = 0; t < 4; ++t) {
      int vc = t * 16 + col;
      afr[t] = a_frag(Vv + ((size_t)b * 64 + vc) * NPIX, pb, half);
      int kc = t * 16 + col;
      bfr[t] = *(const v16bf*)(Kn + ((size_t)b * 64 + kc) * NPIX + pb + half * 16);
#pragma unroll
      for (int e = 0; e < 16; ++e) {
        vs[t] += (float)afr[t][e];
        ksm[t] += (float)bfr[t][e];
      }
    }
#pragma unroll
    for (int vt = 0; vt < 4; ++vt)
#pragma unroll
      for (int kt = 0; kt < 4; ++kt)
        acc[vt][kt] = wmma_bf16(afr[vt], bfr[kt], acc[vt][kt]);
  }

  // block-level reduction in LDS (ds atomics), then single global flush
#pragma unroll
  for (int t = 0; t < 4; ++t) {
    float v = vs[t] + __shfl_xor(vs[t], 16, 32);
    float k = ksm[t] + __shfl_xor(ksm[t], 16, 32);
    if (half == 0) {
      atomicAdd(&vred[t * 16 + col], v);
      atomicAdd(&kred[t * 16 + col], k);
    }
  }
#pragma unroll
  for (int vt = 0; vt < 4; ++vt)
#pragma unroll
    for (int kt = 0; kt < 4; ++kt)
#pragma unroll
      for (int r = 0; r < 8; ++r)
        atomicAdd(&mred[vt * 16 + half * 8 + r][kt * 16 + col], acc[vt][kt][r]);
  __syncthreads();
  for (int i = tid; i < 4096; i += 256)
    atomicAdd(&MT[(size_t)b * 4096 + i], mred[i >> 6][i & 63]);
  if (tid < 64) {
    atomicAdd(&Ksum[b * 64 + tid], kred[tid]);
    atomicAdd(&Vsum[b * 64 + tid], vred[tid]);
  }
}

// ---------------------------------------------------------------------------
// Kernel 3: T = MT * Qn ; numer = Vsum + T ; wv = numer * denom ; att = Wr*wv+br
//   att written as (B,H,W,C) bf16 (aliases Qn storage; reads precede writes)
// ---------------------------------------------------------------------------
__global__ __launch_bounds__(256) void apply_kernel(
    const bf16* Qn, const float* __restrict__ MT,
    const float* __restrict__ Ksum, const float* __restrict__ Vsum,
    const float* __restrict__ Wr, const float* __restrict__ br, bf16* att) {
  __shared__ bf16 Ml[64][64];
  __shared__ bf16 Wrl[64][64];
  __shared__ float ksl[64], vsl[64], brl[64];
  __shared__ bf16 wv[8][16][64];

  int tid = threadIdx.x;
  long g0 = (long)blockIdx.x * 128;
  int b = (int)(g0 >> 16);
  for (int i = tid; i < 4096; i += 256) {
    Ml[i >> 6][i & 63] = (bf16)MT[(size_t)b * 4096 + i];
    Wrl[i >> 6][i & 63] = (bf16)Wr[i];
  }
  for (int i = tid; i < 64; i += 256) {
    ksl[i] = Ksum[b * 64 + i] + 1e-6f;
    vsl[i] = Vsum[b * 64 + i];
    brl[i] = br[i];
  }
  __syncthreads();

  int wave = tid >> 5, lane = tid & 31, half = lane >> 4, col = lane & 15;
  int p = (int)(g0 & (NPIX - 1)) + wave * 16 + col;
  size_t qbase = ((size_t)b * NPIX + p) * 64;

  v16bf qb[2];
  float dd = 0.f;
#pragma unroll
  for (int ks = 0; ks < 2; ++ks) {
    int c0 = ks * 32 + half * 16;
    qb[ks] = *(const v16bf*)(Qn + qbase + c0);
#pragma unroll
    for (int e = 0; e < 16; ++e) dd += (float)qb[ks][e] * ksl[c0 + e];
  }
  dd += __shfl_xor(dd, 16, 32);
  float dinv = 1.f / ((float)NPIX + dd);

  v8f acc[4];
#pragma unroll
  for (int t = 0; t < 4; ++t)
#pragma unroll
    for (int r = 0; r < 8; ++r) acc[t][r] = vsl[t * 16 + half * 8 + r];
#pragma unroll
  for (int t = 0; t < 4; ++t)
#pragma unroll
    for (int ks = 0; ks < 2; ++ks)
      acc[t] = wmma_bf16(a_frag(&Ml[t * 16 + col][0], ks * 32, half), qb[ks], acc[t]);
#pragma unroll
  for (int t = 0; t < 4; ++t)
#pragma unroll
    for (int r = 0; r < 8; ++r) acc[t][r] *= dinv;

  // route wv through LDS to reshape D-layout -> B-fragment layout
#pragma unroll
  for (int t = 0; t < 4; ++t) {
    v8bf pk;
#pragma unroll
    for (int r = 0; r < 8; ++r) pk[r] = (bf16)acc[t][r];
    *(v8bf*)(&wv[wave][col][t * 16 + half * 8]) = pk;
  }
  __syncthreads();

  v8f acc2[4];
#pragma unroll
  for (int t = 0; t < 4; ++t)
#pragma unroll
    for (int r = 0; r < 8; ++r) acc2[t][r] = brl[t * 16 + half * 8 + r];
  v16bf wb[2];
#pragma unroll
  for (int ks = 0; ks < 2; ++ks)
    wb[ks] = *(const v16bf*)(&wv[wave][col][ks * 32 + half * 16]);
#pragma unroll
  for (int t = 0; t < 4; ++t)
#pragma unroll
    for (int ks = 0; ks < 2; ++ks)
      acc2[t] = wmma_bf16(a_frag(&Wrl[t * 16 + col][0], ks * 32, half), wb[ks], acc2[t]);

#pragma unroll
  for (int t = 0; t < 4; ++t) {
    v8bf pk;
#pragma unroll
    for (int r = 0; r < 8; ++r) pk[r] = (bf16)acc2[t][r];
    *(v8bf*)(att + qbase + t * 16 + half * 8) = pk;
  }
}

// ---------------------------------------------------------------------------
// Kernel 4/5: 3x3 SAME conv as implicit GEMM (K = 9 taps x 64 ch).
//   Weight slices staged in LDS per dy (block-uniform row).
//   final_residual=0: out_hwc bf16 ;  =1: out = conv*x + x  (NCHW f32)
// ---------------------------------------------------------------------------
__global__ __launch_bounds__(256) void conv3x3_kernel(
    const bf16* __restrict__ in, const bf16* __restrict__ Wt,
    const float* __restrict__ bias, bf16* __restrict__ out_hwc,
    const float* __restrict__ x_nchw, float* __restrict__ out_final,
    int final_residual) {
  __shared__ bf16 Wsl[3][64][64];  // 3 dx-slices for current dy
  int tid = threadIdx.x, wave = tid >> 5, lane = tid & 31;
  int half = lane >> 4, col = lane & 15;
  long g0 = (long)blockIdx.x * 128 + wave * 16;
  int b = (int)(g0 >> 16);
  int n = (int)(g0 & (NPIX - 1));
  int h = n >> 8, w0 = n & 255;  // block-uniform h
  int px = w0 + col;

  v8f acc[4];
#pragma unroll
  for (int t = 0; t < 4; ++t)
#pragma unroll
    for (int r = 0; r < 8; ++r) acc[t][r] = bias[t * 16 + half * 8 + r];

  for (int dy = 0; dy < 3; ++dy) {
    __syncthreads();  // protect previous slice reads before overwrite
    {
      const v8bf* src = (const v8bf*)(Wt + (size_t)dy * 3 * 4096);
      v8bf* dst = (v8bf*)&Wsl[0][0][0];
      for (int i = tid; i < 1536; i += 256) dst[i] = src[i];
    }
    __syncthreads();

    int hy = h + dy - 1;
    bool row_ok = (hy >= 0) && (hy < 256);
#pragma unroll
    for (int dx = 0; dx < 3; ++dx) {
      int wx = px + dx - 1;
      bool ok = row_ok && (wx >= 0) && (wx < 256);
      const bf16* ip = in + (((size_t)b * NPIX + (size_t)hy * 256 + wx) * 64);
      v16bf bfrag[2];
#pragma unroll
      for (int ks = 0; ks < 2; ++ks) {
        v16bf z = zero_frag();
        if (ok) z = *(const v16bf*)(ip + ks * 32 + half * 16);
        bfrag[ks] = z;
      }
#pragma unroll
      for (int t = 0; t < 4; ++t) {
        int o = t * 16 + col;
#pragma unroll
        for (int ks = 0; ks < 2; ++ks)
          acc[t] = wmma_bf16(a_frag(&Wsl[dx][o][0], ks * 32, half), bfrag[ks], acc[t]);
      }
    }
  }

  if (!final_residual) {
    size_t obase = ((size_t)b * NPIX + n + col) * 64;
#pragma unroll
    for (int t = 0; t < 4; ++t) {
      v8bf pk;
#pragma unroll
      for (int r = 0; r < 8; ++r) pk[r] = (bf16)acc[t][r];
      *(v8bf*)(out_hwc + obase + t * 16 + half * 8) = pk;
    }
  } else {
#pragma unroll
    for (int t = 0; t < 4; ++t)
#pragma unroll
      for (int r = 0; r < 8; ++r) {
        int c = t * 16 + half * 8 + r;
        size_t o = ((size_t)b * 64 + c) * NPIX + n + col;
        float xv = x_nchw[o];
        out_final[o] = fmaf(acc[t][r], xv, xv);
      }
  }
}

// ---------------------------------------------------------------------------
extern "C" void kernel_launch(void* const* d_in, const int* in_sizes, int n_in,
                              void* d_out, int out_size, void* d_ws, size_t ws_size,
                              hipStream_t stream) {
  const float* x  = (const float*)d_in[0];
  const float* Wq = (const float*)d_in[1];
  const float* bq = (const float*)d_in[2];
  const float* Wk = (const float*)d_in[3];
  const float* bk = (const float*)d_in[4];
  const float* Wv = (const float*)d_in[5];
  const float* bv = (const float*)d_in[6];
  const float* Wr = (const float*)d_in[7];
  const float* br = (const float*)d_in[8];
  const float* W1 = (const float*)d_in[9];
  const float* b1 = (const float*)d_in[10];
  const float* W2 = (const float*)d_in[11];
  const float* b2 = (const float*)d_in[12];
  float* out = (float*)d_out;

  char* ws = (char*)d_ws;
  const size_t SZ_BNC = (size_t)BATCH * NPIX * 64 * sizeof(bf16); // 64 MB
  bf16*  QN  = (bf16*)(ws);
  bf16*  KN  = (bf16*)(ws + SZ_BNC);
  bf16*  VV  = (bf16*)(ws + 2 * SZ_BNC);
  float* MT  = (float*)(ws + 3 * SZ_BNC);                    // 8*64*64 f32
  float* KS  = (float*)(ws + 3 * SZ_BNC + 131072);           // 8*64 f32
  float* VS  = (float*)(ws + 3 * SZ_BNC + 131072 + 2048);    // 8*64 f32
  bf16*  W1T = (bf16*)(ws + 3 * SZ_BNC + 131072 + 4096);     // 9*64*64 bf16
  bf16*  W2T = (bf16*)(ws + 3 * SZ_BNC + 131072 + 4096 + 73728);
  bf16*  ATT = QN;  // reuse: each wave reads its Qn pixels before writing att
  bf16*  H1  = KN;  // Kn fully consumed by kv_reduce before conv1 runs

  // zero region = MT|KS|VS contiguous (33792 floats); prep also transforms W1/W2
  prep_kernel<<<dim3(420), dim3(256), 0, stream>>>(W1, W2, W1T, W2T, MT);
  qkv_kernel<<<dim3(4096), dim3(256), 0, stream>>>(x, Wq, bq, Wk, bk, Wv, bv,
                                                   QN, KN, VV);
  kv_reduce_kernel<<<dim3(256), dim3(256), 0, stream>>>(KN, VV, MT, KS, VS);
  apply_kernel<<<dim3(4096), dim3(256), 0, stream>>>(QN, MT, KS, VS, Wr, br, ATT);
  conv3x3_kernel<<<dim3(4096), dim3(256), 0, stream>>>(ATT, W1T, b1, H1,
                                                       nullptr, nullptr, 0);
  conv3x3_kernel<<<dim3(4096), dim3(256), 0, stream>>>(H1, W2T, b2, nullptr,
                                                       x, out, 1);
}
